// ContentionPredictor_53300544143409
// MI455X (gfx1250) — compile-verified, matrix operand order checked by reading, not run on hardware
//
#include <hip/hip_runtime.h>

typedef __attribute__((ext_vector_type(16))) _Float16 v16h;
typedef __attribute__((ext_vector_type(8)))  float    v8f;
typedef __attribute__((ext_vector_type(2)))  float    v2f;

#define NB   131072   // batch rows
#define DD   1029     // x row stride
#define TT   1024     // tasks / GEMM K
#define NH1  64
#define NH2  32

__device__ __forceinline__ v8f wmma_f16(v16h a, v16h b, v8f c) {
  // D = A(16x32 f16) * B(32x16 f16) + C(16x16 f32)
  return __builtin_amdgcn_wmma_f32_16x16x32_f16(false, a, false, b, (short)0, c,
                                                false, false);
}
__device__ __forceinline__ v8f wmma_f32(v2f a, v2f b, v8f c) {
  // D = A(16x4 f32) * B(4x16 f32) + C(16x16 f32)
  return __builtin_amdgcn_wmma_f32_16x16x4_f32(false, a, false, b, (short)0, c,
                                               false, false);
}

// ---------------------------------------------------------------------------
// rowmean[t] = mean(contention[t, :])
// ---------------------------------------------------------------------------
__global__ __launch_bounds__(256) void cp_rowmean(const float* __restrict__ cont,
                                                  float* __restrict__ rowmean) {
  __shared__ float red[256];
  const int t = blockIdx.x;
  const float* row = cont + (size_t)t * TT;
  float s = 0.f;
  for (int j = threadIdx.x; j < TT; j += 256) s += row[j];
  red[threadIdx.x] = s;
  __syncthreads();
  for (int off = 128; off > 0; off >>= 1) {
    if (threadIdx.x < off) red[threadIdx.x] += red[threadIdx.x + off];
    __syncthreads();
  }
  if (threadIdx.x == 0) rowmean[t] = red[0] * (1.0f / TT);
}

// ---------------------------------------------------------------------------
// Pack w1 rows 5..1028 into f16 hi/lo copies laid out in the
// V_WMMA_F32_16X16X32_F16 B-matrix (32x16, K-major pairs) lane/VGPR order.
// dword index = ((((hl*4 + nt)*32 + kc)*32) + lane)*8 + r
// ---------------------------------------------------------------------------
__global__ __launch_bounds__(256) void cp_pack_w1(const float* __restrict__ w1,
                                                  unsigned int* __restrict__ pk) {
  const int idx = blockIdx.x * 256 + threadIdx.x;
  if (idx >= 2 * 4 * 32 * 32 * 8) return;
  const int r    = idx & 7;
  const int lane = (idx >> 3) & 31;
  const int kc   = (idx >> 8) & 31;
  const int nt   = (idx >> 13) & 3;
  const int hl   = (idx >> 15) & 1;
  const int half = lane >> 4;
  const int n    = nt * 16 + (lane & 15);
  const int kin  = (r < 4) ? (8 * half + 2 * r) : (16 + 8 * half + 2 * (r - 4));
  const int k0   = kc * 32 + kin;
  const float v0 = w1[(size_t)(5 + k0) * NH1 + n];
  const float v1 = w1[(size_t)(5 + k0 + 1) * NH1 + n];
  _Float16 h0, h1;
  if (hl == 0) {
    h0 = (_Float16)v0;
    h1 = (_Float16)v1;
  } else {
    h0 = (_Float16)(v0 - (float)(_Float16)v0);
    h1 = (_Float16)(v1 - (float)(_Float16)v1);
  }
  unsigned short u0, u1;
  __builtin_memcpy(&u0, &h0, 2);
  __builtin_memcpy(&u1, &h1, 2);
  pk[idx] = (unsigned int)u0 | ((unsigned int)u1 << 16);
}

// ---------------------------------------------------------------------------
// Main fused kernel.
// ---------------------------------------------------------------------------
__global__ __launch_bounds__(256) void cp_main(
    const float* __restrict__ x, const float* __restrict__ rowmean,
    const uint4* __restrict__ w1pack, const float* __restrict__ w1,
    const float* __restrict__ b1, const float* __restrict__ w2,
    const float* __restrict__ b2, const float* __restrict__ w3,
    const float* __restrict__ b3, float* __restrict__ out) {
  __shared__ __align__(16) float s_w2t[NH2][68];    // w2 transposed, padded
  __shared__ __align__(16) float s_b2[NH2];
  __shared__ __align__(16) float s_w3[NH2];
  __shared__ float s_g[8][16][6];                   // feats+ce per row
  __shared__ __align__(16) float s_h1[8][16][68];   // padded h1 staging

  const int tid  = threadIdx.x;
  const int lane = tid & 31;
  const int wave = tid >> 5;
  const int m16  = lane & 15;
  const int half = lane >> 4;

  for (int i = tid; i < NH1 * NH2; i += 256) s_w2t[i & 31][i >> 5] = w2[i];
  if (tid < NH2) { s_b2[tid] = b2[tid]; s_w3[tid] = w3[tid]; }
  const float b3v = b3[0];
  __syncthreads();

  const size_t rowBase = ((size_t)blockIdx.x * 8 + wave) * 32;
  const float* xr0 = x + (rowBase + m16) * (size_t)DD + 4;        // tile 0 row
  const float* xr1 = x + (rowBase + 16 + m16) * (size_t)DD + 4;   // tile 1 row

  v8f acc[2][4];
#pragma unroll
  for (int t = 0; t < 2; ++t)
#pragma unroll
    for (int nt = 0; nt < 4; ++nt) { v8f z = {}; acc[t][nt] = z; }

  float best0 = -3.402823466e38f, best1 = -3.402823466e38f;
  int bkc0 = 0, bkc1 = 0;  // winning 32-element K-chunk per lane

  for (int kc = 0; kc < 32; ++kc) {
    const int kb = kc * 32;
    v16h ah0, al0, ah1, al1;
    // ---- A tile 0: load fp32, split hi/lo, chunk-max only ----
    {
      const float* p0 = xr0 + kb + 8 * half;
      float fv[16];
#pragma unroll
      for (int i = 0; i < 8; ++i) { fv[i] = p0[i]; fv[8 + i] = p0[16 + i]; }
#pragma unroll
      for (int i = 0; i < 16; ++i) {
        _Float16 h = (_Float16)fv[i];
        ah0[i] = h; al0[i] = (_Float16)(fv[i] - (float)h);
      }
      float a0 = fmaxf(fv[0], fv[1]),   a1 = fmaxf(fv[2], fv[3]);
      float a2 = fmaxf(fv[4], fv[5]),   a3 = fmaxf(fv[6], fv[7]);
      float a4 = fmaxf(fv[8], fv[9]),   a5 = fmaxf(fv[10], fv[11]);
      float a6 = fmaxf(fv[12], fv[13]), a7 = fmaxf(fv[14], fv[15]);
      float m = fmaxf(fmaxf(fmaxf(a0, a1), fmaxf(a2, a3)),
                      fmaxf(fmaxf(a4, a5), fmaxf(a6, a7)));
      if (m > best0) { best0 = m; bkc0 = kc; }
    }
    // ---- A tile 1 ----
    {
      const float* p0 = xr1 + kb + 8 * half;
      float fv[16];
#pragma unroll
      for (int i = 0; i < 8; ++i) { fv[i] = p0[i]; fv[8 + i] = p0[16 + i]; }
#pragma unroll
      for (int i = 0; i < 16; ++i) {
        _Float16 h = (_Float16)fv[i];
        ah1[i] = h; al1[i] = (_Float16)(fv[i] - (float)h);
      }
      float a0 = fmaxf(fv[0], fv[1]),   a1 = fmaxf(fv[2], fv[3]);
      float a2 = fmaxf(fv[4], fv[5]),   a3 = fmaxf(fv[6], fv[7]);
      float a4 = fmaxf(fv[8], fv[9]),   a5 = fmaxf(fv[10], fv[11]);
      float a6 = fmaxf(fv[12], fv[13]), a7 = fmaxf(fv[14], fv[15]);
      float m = fmaxf(fmaxf(fmaxf(a0, a1), fmaxf(a2, a3)),
                      fmaxf(fmaxf(a4, a5), fmaxf(a6, a7)));
      if (m > best1) { best1 = m; bkc1 = kc; }
    }
    // ---- B tiles (prepacked, L2-hot) + 6 WMMAs per n-tile ----
#pragma unroll
    for (int nt = 0; nt < 4; ++nt) {
      union { uint4 q[2]; v16h h; } bh, bl;
      const int ih = ((nt * 32 + kc) * 32 + lane) * 2;
      const int il = (((4 + nt) * 32 + kc) * 32 + lane) * 2;
      bh.q[0] = w1pack[ih]; bh.q[1] = w1pack[ih + 1];
      bl.q[0] = w1pack[il]; bl.q[1] = w1pack[il + 1];
      acc[0][nt] = wmma_f16(ah0, bh.h, acc[0][nt]);
      acc[1][nt] = wmma_f16(ah1, bh.h, acc[1][nt]);
      acc[0][nt] = wmma_f16(al0, bh.h, acc[0][nt]);
      acc[1][nt] = wmma_f16(al1, bh.h, acc[1][nt]);
      acc[0][nt] = wmma_f16(ah0, bl.h, acc[0][nt]);
      acc[1][nt] = wmma_f16(ah1, bl.h, acc[1][nt]);
    }
  }

  // ---- recover exact first-index argmax: rescan the winning 32-elem chunk
  //      (descending k, == compare -> smallest k wins) ----
  int bk0, bk1;
  {
    const float* q0 = xr0 + bkc0 * 32 + 8 * half;
    int bi = 0;
#pragma unroll
    for (int i = 7; i >= 0; --i)
      if (q0[16 + i] == best0) bi = bkc0 * 32 + 16 + 8 * half + i;
#pragma unroll
    for (int i = 7; i >= 0; --i)
      if (q0[i] == best0) bi = bkc0 * 32 + 8 * half + i;
    bk0 = bi;
  }
  {
    const float* q0 = xr1 + bkc1 * 32 + 8 * half;
    int bi = 0;
#pragma unroll
    for (int i = 7; i >= 0; --i)
      if (q0[16 + i] == best1) bi = bkc1 * 32 + 16 + 8 * half + i;
#pragma unroll
    for (int i = 7; i >= 0; --i)
      if (q0[i] == best1) bi = bkc1 * 32 + 8 * half + i;
    bk1 = bi;
  }
  // combine across the lane pair holding each row (value, then lower index)
  {
    float ob = __shfl_xor(best0, 16, 32);
    int   ok = __shfl_xor(bk0, 16, 32);
    if (ob > best0 || (ob == best0 && ok < bk0)) { best0 = ob; bk0 = ok; }
    ob = __shfl_xor(best1, 16, 32);
    ok = __shfl_xor(bk1, 16, 32);
    if (ob > best1 || (ob == best1 && ok < bk1)) { best1 = ob; bk1 = ok; }
  }

#pragma unroll
  for (int t = 0; t < 2; ++t) {
    const int bkk = (t == 0) ? bk0 : bk1;
    __syncthreads();  // s_g / s_h1 reuse across tiles
    if (half == 0) {  // lanes 0..15 own row (rowBase + t*16 + m16)
      const float* rp = x + (rowBase + t * 16 + m16) * (size_t)DD;
      const float d0 = rp[0];
      s_g[wave][m16][0] = d0;
      s_g[wave][m16][1] = rp[1];
      s_g[wave][m16][2] = rp[2];
      s_g[wave][m16][3] = rp[3];
      s_g[wave][m16][4] = rowmean[bkk] * d0;  // contention effect
    }
    __syncthreads();

    // rank-5 correction + bias + relu; scatter C-layout acc into LDS h1
    float w1c[4][6];
#pragma unroll
    for (int nt = 0; nt < 4; ++nt) {
      const int n = nt * 16 + m16;
      w1c[nt][0] = w1[n];
      w1c[nt][1] = w1[NH1 + n];
      w1c[nt][2] = w1[2 * NH1 + n];
      w1c[nt][3] = w1[3 * NH1 + n];
      w1c[nt][4] = w1[4 * NH1 + n];
      w1c[nt][5] = b1[n];
    }
#pragma unroll
    for (int r = 0; r < 8; ++r) {
      const int M = r + 8 * half;  // C/D layout: VGPR r -> M = r + 8*(lane/16)
      const float g0 = s_g[wave][M][0], g1 = s_g[wave][M][1];
      const float g2 = s_g[wave][M][2], g3 = s_g[wave][M][3];
      const float g4 = s_g[wave][M][4];
#pragma unroll
      for (int nt = 0; nt < 4; ++nt) {
        float hsum = acc[t][nt][r] + w1c[nt][5];
        hsum = fmaf(g0, w1c[nt][0], hsum);
        hsum = fmaf(g1, w1c[nt][1], hsum);
        hsum = fmaf(g2, w1c[nt][2], hsum);
        hsum = fmaf(g3, w1c[nt][3], hsum);
        hsum = fmaf(g4, w1c[nt][4], hsum);
        s_h1[wave][M][nt * 16 + m16] = fmaxf(hsum, 0.f);
      }
    }
    __syncthreads();

    // ---- layer 2 via V_WMMA_F32_16X16X4_F32: H2 = H1(16x64) @ W2(64x32) ----
    v8f d2[2];
#pragma unroll
    for (int nt = 0; nt < 2; ++nt) { v8f z = {}; d2[nt] = z; }
#pragma unroll
    for (int ks = 0; ks < 16; ++ks) {
      const int k0 = ks * 4 + 2 * half;  // A: lane supplies K = k0, k0+1
      const v2f aa = *(const v2f*)&s_h1[wave][m16][k0];
#pragma unroll
      for (int nt = 0; nt < 2; ++nt) {
        const v2f bb = *(const v2f*)&s_w2t[nt * 16 + m16][k0];
        d2[nt] = wmma_f32(aa, bb, d2[nt]);
      }
    }
    // relu(h2 + b2) . w3, reduced over N via butterfly within each 16-lane half
    float c[8];
#pragma unroll
    for (int r = 0; r < 8; ++r) {
      float s = 0.f;
#pragma unroll
      for (int nt = 0; nt < 2; ++nt) {
        const int n = nt * 16 + m16;
        const float hb = fmaxf(d2[nt][r] + s_b2[n], 0.f);
        s = fmaf(hb, s_w3[n], s);
      }
      c[r] = s;
    }
#pragma unroll
    for (int off = 1; off <= 8; off <<= 1)
#pragma unroll
      for (int r = 0; r < 8; ++r) c[r] += __shfl_xor(c[r], off, 32);
    if (m16 == 0) {  // lane 0 -> rows M=0..7, lane 16 -> rows M=8..15
#pragma unroll
      for (int r = 0; r < 8; ++r)
        out[rowBase + t * 16 + r + 8 * half] = fmaxf(c[r] + b3v, 0.f);
    }
  }
}

// ---------------------------------------------------------------------------
extern "C" void kernel_launch(void* const* d_in, const int* in_sizes, int n_in,
                              void* d_out, int out_size, void* d_ws,
                              size_t ws_size, hipStream_t stream) {
  (void)in_sizes; (void)n_in; (void)out_size; (void)ws_size;
  const float* x    = (const float*)d_in[0];
  const float* cont = (const float*)d_in[1];
  const float* w1   = (const float*)d_in[2];
  const float* b1   = (const float*)d_in[3];
  const float* w2   = (const float*)d_in[4];
  const float* b2   = (const float*)d_in[5];
  const float* w3   = (const float*)d_in[6];
  const float* b3   = (const float*)d_in[7];
  float* out = (float*)d_out;

  float* rowmean = (float*)d_ws;                              // 4 KB
  unsigned int* w1pk = (unsigned int*)((char*)d_ws + 4096);   // 256 KB packed

  cp_rowmean<<<TT, 256, 0, stream>>>(cont, rowmean);
  cp_pack_w1<<<(2 * 4 * 32 * 32 * 8) / 256, 256, 0, stream>>>(w1, w1pk);
  cp_main<<<NB / 256, 256, 0, stream>>>(x, rowmean, (const uint4*)w1pk, w1, b1,
                                        w2, b2, w3, b3, out);
}
